// LinearAttention_5669356834077
// MI455X (gfx1250) — matrix operand order repaired and checked by reading, not compiled, not run on hardware
//
#include <hip/hip_runtime.h>
#include <math.h>

// ---------------------------------------------------------------------------
// LinearAttention on MI455X (gfx1250, wave32, WMMA).
//
// All five GEMMs use v_wmma_f32_16x16x32_bf16 (fp32 accumulate).
// Operand layouts are chosen so WMMA fragments are direct per-lane loads:
//   A (16x32 bf16): lane l holds row M=l%16, K = (l/16)*16+[0..16)
//                   -> two 16B loads from a K-contiguous row
//   B (32x16 bf16): lane l holds col N=l%16, K = (l/16)*16+[0..16)
//                   -> one 32B load from a K-contiguous "column" (row of Bt)
//   C/D (16x16 f32): VGPR r, lane l -> (M = r + 8*(l/16), N = l%16)
//
// All K-loops are double-buffered: fragments for step i+1 are loaded before
// the WMMAs of step i so the scheduler can issue partial s_wait_loadcnt
// instead of serializing load->wmma every step.
// ---------------------------------------------------------------------------

#define B_ 16
#define DIM_ 256
#define H_ 64
#define W_ 64
#define N_ 4096            // H*W tokens per image
#define NPIX 65536         // B * N_
#define HEADS_ 8
#define DH_ 64
#define INNER_ 512
#define BH_ 128            // B * HEADS

typedef __attribute__((ext_vector_type(16))) __bf16 bf16x16;
typedef __attribute__((ext_vector_type(8)))  __bf16 bf16x8;
typedef __attribute__((ext_vector_type(8)))  float  v8f;

union AFrag { bf16x16 v; bf16x8 h[2]; };

#define V8Z {0.f,0.f,0.f,0.f,0.f,0.f,0.f,0.f}

__device__ __forceinline__ v8f wmma_bf16(bf16x16 a, bf16x16 b, v8f c) {
  // D = A(16x32) * B(32x16) + C, fp32 accumulate
  return __builtin_amdgcn_wmma_f32_16x16x32_bf16(false, a, false, b,
                                                 (short)0, c, false, false);
}

__device__ __forceinline__ void load_a(AFrag& a, const __bf16* p) {
  a.h[0] = *(const bf16x8*)(p);
  a.h[1] = *(const bf16x8*)(p + 8);
}

// ---------------------------------------------------------------- kernel 1 --
// fp32 weights -> bf16 (row-major kept: A operands read rows K-contiguous)
__global__ __launch_bounds__(256) void k_wconv(
    const float* __restrict__ wq, const float* __restrict__ wpw,
    const float* __restrict__ wout,
    __bf16* __restrict__ wq_b, __bf16* __restrict__ wpw_b,
    __bf16* __restrict__ wout_b) {
  const int i = blockIdx.x * 256 + threadIdx.x;
  if (i < INNER_ * DIM_)     wq_b[i]   = (__bf16)wq[i];
  if (i < 2 * INNER_ * DIM_) wpw_b[i]  = (__bf16)wpw[i];
  if (i < DIM_ * INNER_)     wout_b[i] = (__bf16)wout[i];
}

// ---------------------------------------------------------------- kernel 2 --
// depthwise 3x3 (zero pad) + cast; LDS 16x16 transpose so outputs are
// pixel-major bf16: fmap_t[p][c], dw_t[p][c]  (B operands for GEMM 1/2).
__global__ __launch_bounds__(256) void k_dwconv(
    const float* __restrict__ fmap, const float* __restrict__ Wdw,
    __bf16* __restrict__ fmap_t, __bf16* __restrict__ dw_t) {
  __shared__ float tf[16][17], td[16][17];
  const int tx = threadIdx.x, ty = threadIdx.y;
  const int p = blockIdx.x * 16 + tx;        // pixel (coalesced reads)
  const int c = blockIdx.y * 16 + ty;        // channel
  const int b = p >> 12, n = p & (N_ - 1), h = n >> 6, w = n & 63;
  const float* src = fmap + ((size_t)(b * DIM_ + c) << 12);
  float acc = 0.f;
  #pragma unroll
  for (int dy = -1; dy <= 1; ++dy)
    #pragma unroll
    for (int dx = -1; dx <= 1; ++dx) {
      const int hh = h + dy, ww = w + dx;
      if (hh >= 0 && hh < H_ && ww >= 0 && ww < W_)
        acc += Wdw[c * 9 + (dy + 1) * 3 + (dx + 1)] * src[hh * W_ + ww];
    }
  tf[ty][tx] = src[n];
  td[ty][tx] = acc;
  __syncthreads();
  const int op = blockIdx.x * 16 + ty;       // pixel (now in ty)
  const int oc = blockIdx.y * 16 + tx;       // channel (coalesced writes)
  fmap_t[(size_t)op * DIM_ + oc] = (__bf16)tf[tx][ty];
  dw_t  [(size_t)op * DIM_ + oc] = (__bf16)td[tx][ty];
}

// ---------------------------------------------------------------- kernel 3 --
// q = Wq * x, one wave owns a full head (64 M) x 16 pixels so softmax over
// dh happens in-register (lane l and l^16 share column N=l%16).
__global__ __launch_bounds__(256) void k_qgemm_softmax(
    const __bf16* __restrict__ Wq, const __bf16* __restrict__ Xt,
    __bf16* __restrict__ q_s) {
  const int wid  = blockIdx.x * 8 + (threadIdx.x >> 5);
  const int lane = threadIdx.x & 31, lm = lane & 15, lh = lane >> 4;
  const int head  = wid >> 12;               // 0..7
  const int pBase = (wid & 4095) * 16;       // pixel tile
  const int mBase = head * DH_;
  v8f acc[4] = {V8Z, V8Z, V8Z, V8Z};
  const __bf16* brow = Xt + (size_t)(pBase + lm) * DIM_ + lh * 16;
  const __bf16* ar0  = Wq + (size_t)(mBase + lm) * DIM_ + lh * 16;
  // prologue loads for k0 = 0
  bf16x16 bv = *(const bf16x16*)(brow);
  AFrag a[4];
  #pragma unroll
  for (int t = 0; t < 4; ++t) load_a(a[t], ar0 + (size_t)t * 16 * DIM_);
  #pragma unroll
  for (int k0 = 0; k0 < DIM_; k0 += 32) {
    const int k1 = (k0 + 32) & (DIM_ - 1);   // wrap: epilogue prefetch is benign
    bf16x16 bvn = *(const bf16x16*)(brow + k1);
    AFrag an[4];
    #pragma unroll
    for (int t = 0; t < 4; ++t) load_a(an[t], ar0 + (size_t)t * 16 * DIM_ + k1);
    #pragma unroll
    for (int t = 0; t < 4; ++t) acc[t] = wmma_bf16(a[t].v, bv, acc[t]);
    bv = bvn;
    #pragma unroll
    for (int t = 0; t < 4; ++t) a[t] = an[t];
  }
  // softmax over the 64 dh values of this head, per pixel column
  float mx = -3.4e38f;
  #pragma unroll
  for (int t = 0; t < 4; ++t)
    #pragma unroll
    for (int r = 0; r < 8; ++r) mx = fmaxf(mx, acc[t][r]);
  mx = fmaxf(mx, __shfl_xor(mx, 16, 32));
  float s = 0.f;
  #pragma unroll
  for (int t = 0; t < 4; ++t)
    #pragma unroll
    for (int r = 0; r < 8; ++r) {
      const float e = expf(acc[t][r] - mx); acc[t][r] = e; s += e;
    }
  s += __shfl_xor(s, 16, 32);
  const float sc = 0.125f / s;               // * DH^-0.5
  const int p = pBase + lm, b = p >> 12, n = p & (N_ - 1);
  __bf16* dst = q_s + ((size_t)(b * HEADS_ + head) * N_ + n) * DH_;
  #pragma unroll
  for (int t = 0; t < 4; ++t)
    #pragma unroll
    for (int r = 0; r < 8; ++r)
      dst[t * 16 + r + 8 * lh] = (__bf16)(acc[t][r] * sc);
}

// ---------------------------------------------------------------- kernel 4 --
// kv = Wpw * dw; epilogue scatters k -> k_raw[bh][d][n] (f32, softmax layout)
// and v -> v_s[bh][e][n] (bf16, K-contiguous B operand for ctx GEMM).
__global__ __launch_bounds__(256) void k_kvgemm(
    const __bf16* __restrict__ Wpw, const __bf16* __restrict__ Dt,
    float* __restrict__ k_raw, __bf16* __restrict__ v_s) {
  const int wid  = blockIdx.x * 8 + (threadIdx.x >> 5);
  const int lane = threadIdx.x & 31, lm = lane & 15, lh = lane >> 4;
  const int mBase = (wid >> 10) * 16;        // 0..1008
  const int pBase = (wid & 1023) * 64;
  v8f acc[4] = {V8Z, V8Z, V8Z, V8Z};
  const __bf16* arow = Wpw + (size_t)(mBase + lm) * DIM_ + lh * 16;
  const __bf16* br0  = Dt + (size_t)(pBase + lm) * DIM_ + lh * 16;
  AFrag a; load_a(a, arow);
  bf16x16 bv[4];
  #pragma unroll
  for (int t = 0; t < 4; ++t) bv[t] = *(const bf16x16*)(br0 + (size_t)t * 16 * DIM_);
  #pragma unroll
  for (int k0 = 0; k0 < DIM_; k0 += 32) {
    const int k1 = (k0 + 32) & (DIM_ - 1);
    AFrag an; load_a(an, arow + k1);
    bf16x16 bn[4];
    #pragma unroll
    for (int t = 0; t < 4; ++t)
      bn[t] = *(const bf16x16*)(br0 + (size_t)t * 16 * DIM_ + k1);
    #pragma unroll
    for (int t = 0; t < 4; ++t) acc[t] = wmma_bf16(a.v, bv[t], acc[t]);
    a = an;
    #pragma unroll
    for (int t = 0; t < 4; ++t) bv[t] = bn[t];
  }
  #pragma unroll
  for (int t = 0; t < 4; ++t) {
    const int p = pBase + t * 16 + lm, b = p >> 12, n = p & (N_ - 1);
    #pragma unroll
    for (int r = 0; r < 8; ++r) {
      const int o = mBase + r + 8 * lh;      // tile never straddles o==512
      if (o < INNER_) {
        const int head = o >> 6, d = o & 63;
        k_raw[((size_t)((b * HEADS_ + head) * DH_ + d)) * N_ + n] = acc[t][r];
      } else {
        const int oe = o - INNER_, head = oe >> 6, e = oe & 63;
        v_s[((size_t)((b * HEADS_ + head) * DH_ + e)) * N_ + n] = (__bf16)acc[t][r];
      }
    }
  }
}

// ---------------------------------------------------------------- kernel 5 --
// token softmax per (bh, d) row of 4096; writes bf16 IN PLACE over the f32
// row (bf16 row pitch = 8192 elems = the 16KB f32 row; only first 4096 used).
__global__ __launch_bounds__(256) void k_ksoftmax(float* __restrict__ k_raw) {
  const int row = blockIdx.x;                // bh*64 + d
  const int tid = threadIdx.x, lane = tid & 31, wv = tid >> 5;
  const float* src = k_raw + (size_t)row * N_;
  float v[16];
  #pragma unroll
  for (int i = 0; i < 16; ++i) v[i] = src[tid + i * 256];
  float mx = -3.4e38f;
  #pragma unroll
  for (int i = 0; i < 16; ++i) mx = fmaxf(mx, v[i]);
  #pragma unroll
  for (int off = 16; off > 0; off >>= 1) mx = fmaxf(mx, __shfl_xor(mx, off, 32));
  __shared__ float smax[8], ssum[8];
  if (lane == 0) smax[wv] = mx;
  __syncthreads();                            // also fences the in-place reads
  mx = smax[0];
  #pragma unroll
  for (int i = 1; i < 8; ++i) mx = fmaxf(mx, smax[i]);
  float s = 0.f;
  #pragma unroll
  for (int i = 0; i < 16; ++i) { v[i] = expf(v[i] - mx); s += v[i]; }
  #pragma unroll
  for (int off = 16; off > 0; off >>= 1) s += __shfl_xor(s, off, 32);
  if (lane == 0) ssum[wv] = s;
  __syncthreads();
  s = ssum[0] + ssum[1] + ssum[2] + ssum[3] +
      ssum[4] + ssum[5] + ssum[6] + ssum[7];
  const float inv = 1.f / s;
  __bf16* dst = (__bf16*)k_raw + (size_t)row * 2 * N_;   // in-place, pitch 8192
  #pragma unroll
  for (int i = 0; i < 16; ++i) dst[tid + i * 256] = (__bf16)(v[i] * inv);
}

// ---------------------------------------------------------------- kernel 6 --
// ctx[d][e] = sum_n k[n,d] v[n,e]; K=4096 (128 steps) -> double-buffered so
// next step's fragments are in flight while this step's WMMAs issue.
// Output stored transposed ctx_t[bh][e][d] bf16 (K-contiguous for kernel 7).
__global__ __launch_bounds__(128) void k_ctxgemm(
    const float* __restrict__ k_mem, const __bf16* __restrict__ v_s,
    __bf16* __restrict__ ctx_t) {
  const int bh = blockIdx.x;
  const int lane = threadIdx.x & 31, lm = lane & 15, lh = lane >> 4;
  const int dBase = (threadIdx.x >> 5) * 16;
  const __bf16* kb = (const __bf16*)k_mem;   // pitch 8192 (in-place bf16)
  v8f acc[4] = {V8Z, V8Z, V8Z, V8Z};
  const __bf16* arow = kb + (size_t)(bh * DH_ + dBase + lm) * (2 * N_) + lh * 16;
  const __bf16* brow = v_s + (size_t)(bh * DH_ + lm) * N_ + lh * 16;
  AFrag a; load_a(a, arow);
  bf16x16 bv[4];
  #pragma unroll
  for (int t = 0; t < 4; ++t) bv[t] = *(const bf16x16*)(brow + (size_t)t * 16 * N_);
  for (int k0 = 0; k0 < N_; k0 += 32) {
    const int k1 = (k0 + 32) & (N_ - 1);     // wrap: final prefetch is benign
    AFrag an; load_a(an, arow + k1);
    bf16x16 bn[4];
    #pragma unroll
    for (int t = 0; t < 4; ++t)
      bn[t] = *(const bf16x16*)(brow + (size_t)t * 16 * N_ + k1);
    #pragma unroll
    for (int t = 0; t < 4; ++t) acc[t] = wmma_bf16(a.v, bv[t], acc[t]);
    a = an;
    #pragma unroll
    for (int t = 0; t < 4; ++t) bv[t] = bn[t];
  }
  #pragma unroll
  for (int t = 0; t < 4; ++t) {
    const int e = t * 16 + lm;
    #pragma unroll
    for (int r = 0; r < 8; ++r) {
      const int d = dBase + r + 8 * lh;
      ctx_t[(size_t)(bh * DH_ + e) * DH_ + d] = (__bf16)acc[t][r];
    }
  }
}

// ---------------------------------------------------------------- kernel 7 --
// out[n][e] = sum_d q_s[n,d] ctx[d,e]; K=64 -> both steps loaded up front.
// Fused exact-erf GELU; writes act_t[pixel][ic] bf16 (ic = head*64+e).
__global__ __launch_bounds__(256) void k_outgemm_gelu(
    const __bf16* __restrict__ q_s, const __bf16* __restrict__ ctx_t,
    __bf16* __restrict__ act_t) {
  const int bh = blockIdx.y, b = bh >> 3, head = bh & 7;
  const int lane = threadIdx.x & 31, lm = lane & 15, lh = lane >> 4;
  const int nBase = blockIdx.x * 128 + (threadIdx.x >> 5) * 16;
  v8f acc[4] = {V8Z, V8Z, V8Z, V8Z};
  const __bf16* arow = q_s + ((size_t)bh * N_ + nBase + lm) * DH_ + lh * 16;
  const __bf16* brow = ctx_t + (size_t)(bh * DH_ + lm) * DH_ + lh * 16;
  AFrag a0, a1;
  load_a(a0, arow);
  load_a(a1, arow + 32);
  bf16x16 b0[4], b1[4];
  #pragma unroll
  for (int t = 0; t < 4; ++t) {
    b0[t] = *(const bf16x16*)(brow + t * 16 * DH_);
    b1[t] = *(const bf16x16*)(brow + t * 16 * DH_ + 32);
  }
  #pragma unroll
  for (int t = 0; t < 4; ++t) acc[t] = wmma_bf16(a0.v, b0[t], acc[t]);
  #pragma unroll
  for (int t = 0; t < 4; ++t) acc[t] = wmma_bf16(a1.v, b1[t], acc[t]);
  #pragma unroll
  for (int t = 0; t < 4; ++t) {
    const int e = t * 16 + lm;
    #pragma unroll
    for (int r = 0; r < 8; ++r) {
      const int n = nBase + r + 8 * lh;
      const float x = acc[t][r];
      const float g = 0.5f * x * (1.f + erff(x * 0.70710678118654752f));
      act_t[((size_t)(b * N_ + n)) * INNER_ + head * DH_ + e] = (__bf16)g;
    }
  }
}

// ---------------------------------------------------------------- kernel 8 --
// y = Wout * act + bout, f32 NCHW output. K=512, double-buffered.
__global__ __launch_bounds__(256) void k_finalgemm(
    const __bf16* __restrict__ Wout, const __bf16* __restrict__ act_t,
    const float* __restrict__ bout, float* __restrict__ out) {
  const int wid  = blockIdx.x * 8 + (threadIdx.x >> 5);
  const int lane = threadIdx.x & 31, lm = lane & 15, lh = lane >> 4;
  const int mBase = (wid >> 10) * 16;        // 0..240
  const int pBase = (wid & 1023) * 64;
  v8f acc[4] = {V8Z, V8Z, V8Z, V8Z};
  const __bf16* arow = Wout + (size_t)(mBase + lm) * INNER_ + lh * 16;
  const __bf16* br0  = act_t + (size_t)(pBase + lm) * INNER_ + lh * 16;
  AFrag a; load_a(a, arow);
  bf16x16 bv[4];
  #pragma unroll
  for (int t = 0; t < 4; ++t) bv[t] = *(const bf16x16*)(br0 + (size_t)t * 16 * INNER_);
  #pragma unroll
  for (int k0 = 0; k0 < INNER_; k0 += 32) {
    const int k1 = (k0 + 32) & (INNER_ - 1);
    AFrag an; load_a(an, arow + k1);
    bf16x16 bn[4];
    #pragma unroll
    for (int t = 0; t < 4; ++t)
      bn[t] = *(const bf16x16*)(br0 + (size_t)t * 16 * INNER_ + k1);
    #pragma unroll
    for (int t = 0; t < 4; ++t) acc[t] = wmma_bf16(a.v, bv[t], acc[t]);
    a = an;
    #pragma unroll
    for (int t = 0; t < 4; ++t) bv[t] = bn[t];
  }
  #pragma unroll
  for (int t = 0; t < 4; ++t) {
    const int p = pBase + t * 16 + lm, b = p >> 12, n = p & (N_ - 1);
    #pragma unroll
    for (int r = 0; r < 8; ++r) {
      const int o = mBase + r + 8 * lh;
      out[(size_t)b * (DIM_ * N_) + (size_t)o * N_ + n] = acc[t][r] + bout[o];
    }
  }
}

// ---------------------------------------------------------------------------
extern "C" void kernel_launch(void* const* d_in, const int* in_sizes, int n_in,
                              void* d_out, int out_size, void* d_ws,
                              size_t ws_size, hipStream_t stream) {
  const float* fmap = (const float*)d_in[0];
  const float* Wq   = (const float*)d_in[1];
  const float* Wdw  = (const float*)d_in[2];
  const float* Wpw  = (const float*)d_in[3];
  const float* Wout = (const float*)d_in[4];
  const float* bout = (const float*)d_in[5];
  float* out = (float*)d_out;

  // workspace carve (~404 MB total), 256B aligned
  char* ws = (char*)d_ws;
  size_t off = 0;
  auto take = [&](size_t bytes) -> char* {
    char* p = ws + off;
    off += (bytes + 255) & ~(size_t)255;
    return p;
  };
  __bf16* fmap_t = (__bf16*)take((size_t)NPIX * DIM_ * 2);        // 32 MB
  __bf16* dw_t   = (__bf16*)take((size_t)NPIX * DIM_ * 2);        // 32 MB
  __bf16* wq_b   = (__bf16*)take((size_t)INNER_ * DIM_ * 2);
  __bf16* wpw_b  = (__bf16*)take((size_t)2 * INNER_ * DIM_ * 2);
  __bf16* wout_b = (__bf16*)take((size_t)DIM_ * INNER_ * 2);
  __bf16* q_s    = (__bf16*)take((size_t)BH_ * N_ * DH_ * 2);     // 64 MB
  float*  k_raw  = (float*) take((size_t)BH_ * DH_ * N_ * 4);     // 128 MB (reused in-place as bf16)
  __bf16* v_s    = (__bf16*)take((size_t)BH_ * DH_ * N_ * 2);     // 64 MB
  __bf16* ctx_t  = (__bf16*)take((size_t)BH_ * DH_ * DH_ * 2);    // 1 MB
  __bf16* act_t  = (__bf16*)take((size_t)NPIX * INNER_ * 2);      // 64 MB

  // 1. weights -> bf16
  k_wconv<<<(2 * INNER_ * DIM_ + 255) / 256, 256, 0, stream>>>(
      Wq, Wpw, Wout, wq_b, wpw_b, wout_b);
  // 2. depthwise 3x3 + transpose to pixel-major bf16
  k_dwconv<<<dim3(NPIX / 16, DIM_ / 16), dim3(16, 16), 0, stream>>>(
      fmap, Wdw, fmap_t, dw_t);
  // 3. q GEMM + fused dh-softmax  (8 heads * 4096 pixel-tiles = 32768 waves)
  k_qgemm_softmax<<<4096, 256, 0, stream>>>(wq_b, fmap_t, q_s);
  // 4. kv GEMM  (64 m-tiles * 1024 n-tiles = 65536 waves)
  k_kvgemm<<<8192, 256, 0, stream>>>(wpw_b, dw_t, k_raw, v_s);
  // 5. k token-softmax (one block per (bh,d) row), bf16 in place
  k_ksoftmax<<<BH_ * DH_, 256, 0, stream>>>(k_raw);
  // 6. ctx = k^T v per bh
  k_ctxgemm<<<BH_, 128, 0, stream>>>(k_raw, v_s, ctx_t);
  // 7. out = q ctx, fused GELU
  k_outgemm_gelu<<<dim3(N_ / 128, BH_), 256, 0, stream>>>(q_s, ctx_t, act_t);
  // 8. final GEMM + bias -> f32 NCHW
  k_finalgemm<<<2048, 256, 0, stream>>>(wout_b, act_t, bout, out);
}
